// ETR_loss_H_58634893525266
// MI455X (gfx1250) — matrix operand order repaired and checked by reading, not compiled, not run on hardware
//
#include <hip/hip_runtime.h>
#include <math.h>

#define NDIM 8192
#define KCOL 128          // columns of C (and H)
#define CHK  256          // K-rows of C staged per LDS chunk
#define LDST (CHK + 8)    // padded column stride (shorts) -> 528B: 132 banks % 64 == 4, conflict-free b128
#define GEMM_LDS_BYTES (KCOL * LDST * 2)

typedef __attribute__((ext_vector_type(16))) __bf16 v16bf;
typedef __attribute__((ext_vector_type(8)))  float  v8f;

union Frag16 {
    v16bf v;
    unsigned int   ui[8];
    uint4 q[2];
};

// two f32 -> packed bf16x2 (round-to-nearest-ish) in 3 VALU ops:
// v_add, v_add, v_perm_b32 picking the high halves ({S0:S1} byte-numbering).
__device__ __forceinline__ unsigned int pk2bf(float lo, float hi) {
    return __builtin_amdgcn_perm(__float_as_uint(hi) + 0x8000u,
                                 __float_as_uint(lo) + 0x8000u,
                                 0x07060302u);
}

// ---------------------------------------------------------------- init
__global__ void etr_init_kernel(float* ws) {
    // ws[0..128) diag, [128..256) colsum, [256] sA, [257] hpart
    for (int i = threadIdx.x; i < 258; i += blockDim.x) ws[i] = 0.0f;
}

// ---------------------------------------------------------------- fused GEMM
// M = A@C with bf16 WMMA; fused: colsum(M), diag(C^T A C), rowsum(A) -> dvec
__global__ __launch_bounds__(256) void etr_gemm_kernel(
        const float* __restrict__ A, const float* __restrict__ C,
        float* __restrict__ diag, float* __restrict__ colsum,
        float* __restrict__ dvec)
{
    __shared__ unsigned short ldsC[KCOL * LDST];   // C chunk, transposed, bf16

    const int tid  = threadIdx.x;
    const int lane = tid & 31;
    const int wave = tid >> 5;
    const int mrow = lane & 15;     // row within 16-row strip
    const int hi   = lane >> 4;     // lane-half selects K sub-offset
    const int rowBase = blockIdx.x * 128 + wave * 16;

    v8f acc[8];
#pragma unroll
    for (int t = 0; t < 8; ++t)
        acc[t] = (v8f){0.f,0.f,0.f,0.f,0.f,0.f,0.f,0.f};

    float rowsum = 0.0f;
    const float* aPtr = A + (size_t)(rowBase + mrow) * NDIM + (hi ? 8 : 0);

    for (int kc = 0; kc < NDIM; kc += CHK) {
        __syncthreads();
        // ---- stage C[kc .. kc+CHK) transposed into LDS as bf16, a K row-pair at a time:
        // rows 2r,2r+1 of column c land in one dword of the column-major LDS image.
#pragma unroll 4
        for (int e = tid; e < (CHK / 2) * KCOL; e += 256) {
            int r2 = e >> 7;          // K row-pair index
            int c  = e & (KCOL - 1);  // column
            const float* src = C + (size_t)(kc + 2 * r2) * KCOL + c;
            unsigned int pk = pk2bf(src[0], src[KCOL]);
            ((unsigned int*)ldsC)[c * (LDST / 2) + r2] = pk;
        }
        __syncthreads();

#pragma unroll 4
        for (int ks = 0; ks < CHK; ks += 32) {
            // ---- A fragment: per-lane K window [c,c+8) U [c+16,c+24), c = base + hi*8
            const float4* ap = (const float4*)aPtr;
            float4 f0 = ap[0], f1 = ap[1];      // K +0..+7
            float4 f2 = ap[4], f3 = ap[5];      // K +16..+23
            __builtin_prefetch(aPtr + 64);      // two steps ahead
            aPtr += 32;

            Frag16 a;
            a.ui[0] = pk2bf(f0.x, f0.y);
            a.ui[1] = pk2bf(f0.z, f0.w);
            a.ui[2] = pk2bf(f1.x, f1.y);
            a.ui[3] = pk2bf(f1.z, f1.w);
            a.ui[4] = pk2bf(f2.x, f2.y);
            a.ui[5] = pk2bf(f2.z, f2.w);
            a.ui[6] = pk2bf(f3.x, f3.y);
            a.ui[7] = pk2bf(f3.z, f3.w);

            rowsum += (f0.x+f0.y+f0.z+f0.w) + (f1.x+f1.y+f1.z+f1.w)
                    + (f2.x+f2.y+f2.z+f2.w) + (f3.x+f3.y+f3.z+f3.w);

            // ---- B fragments from LDS: lane holds column n, K [c,c+8) U [c+16,c+24), c = ks + hi*8
            const int kloc = ks + hi * 8;
#pragma unroll
            for (int t = 0; t < 8; ++t) {
                Frag16 b;
                const uint4* bp = (const uint4*)&ldsC[(t * 16 + mrow) * LDST + kloc];
                b.q[0] = bp[0];
                b.q[1] = bp[2];
                acc[t] = __builtin_amdgcn_wmma_f32_16x16x32_bf16(
                             false, a.v, false, b.v, (short)0, acc[t], false, false);
            }
        }
    }

    // ---- degree (row sum of A): combine lane pair (l, l+16), write once
    float other = __shfl_xor(rowsum, 16, 32);
    if (hi == 0) dvec[rowBase + mrow] = rowsum + other;

    // ---- column sums & diag(C^T A C); D layout: VGPR r <-> M = r + hi*8, lane <-> N
#pragma unroll
    for (int t = 0; t < 8; ++t) {
        float scol = 0.0f, sdiag = 0.0f;
        const int col = t * 16 + mrow;
#pragma unroll
        for (int r = 0; r < 8; ++r) {
            const int row = rowBase + r + hi * 8;
            float m = acc[t][r];
            scol  += m;
            sdiag += m * C[(size_t)row * KCOL + col];
        }
        scol  += __shfl_xor(scol, 16, 32);
        sdiag += __shfl_xor(sdiag, 16, 32);
        if (hi == 0) {
            atomicAdd(&colsum[col], scol);
            atomicAdd(&diag[col],  sdiag);
        }
    }
}

// ---------------------------------------------------------------- h_part & sA
// wave per row: h_part += d_i * ||H_i||^2 ; sA += d_i
__global__ __launch_bounds__(256) void etr_h_kernel(
        const float* __restrict__ H, const float* __restrict__ dvec,
        float* __restrict__ sA, float* __restrict__ hpart)
{
    const int lane = threadIdx.x & 31;
    const int wave = threadIdx.x >> 5;
    const int row  = blockIdx.x * 8 + wave;

    float4 h = ((const float4*)(H + (size_t)row * KCOL))[lane];
    float s = h.x*h.x + h.y*h.y + h.z*h.z + h.w*h.w;
#pragma unroll
    for (int off = 16; off > 0; off >>= 1) s += __shfl_xor(s, off, 32);

    if (lane == 0) {
        float di = dvec[row];
        atomicAdd(hpart, di * s);
        atomicAdd(sA, di);
    }
}

// ---------------------------------------------------------------- finish
__global__ void etr_final_kernel(const float* __restrict__ ws,
                                 const float* __restrict__ papra,
                                 const int* __restrict__ flag,
                                 float* __restrict__ out)
{
    __shared__ float red[128];
    const int t = threadIdx.x;
    const float sA = ws[256];
    red[t] = ws[t] * log2f(ws[128 + t] / sA + 1e-40f);
    __syncthreads();
#pragma unroll
    for (int off = 64; off > 0; off >>= 1) {
        if (t < off) red[t] += red[t + off];
        __syncthreads();
    }
    if (t == 0) {
        float hp = ws[257];
        float total = red[0] / sA;
        if (*flag > 10) total += (*papra) * hp;
        out[0] = total;
        out[1] = hp;
    }
}

// ---------------------------------------------------------------- launcher
extern "C" void kernel_launch(void* const* d_in, const int* in_sizes, int n_in,
                              void* d_out, int out_size, void* d_ws, size_t ws_size,
                              hipStream_t stream) {
    const float* A     = (const float*)d_in[0];
    const float* C     = (const float*)d_in[1];
    const float* H     = (const float*)d_in[2];
    const float* papra = (const float*)d_in[3];
    const int*   flag  = (const int*)d_in[4];
    float* out = (float*)d_out;
    float* ws  = (float*)d_ws;
    // ws layout (floats): [0..128) diag, [128..256) colsum, [256] sA, [257] hpart, [258..258+8192) dvec

    etr_init_kernel <<<1,    256, 0, stream>>>(ws);
    etr_gemm_kernel <<<64,   256, 0, stream>>>(A, C, ws, ws + 128, ws + 258);
    etr_h_kernel    <<<1024, 256, 0, stream>>>(H, ws + 258, ws + 256, ws + 257);
    etr_final_kernel<<<1,    128, 0, stream>>>(ws, papra, flag, out);
}
